// GATActBN_68178310856858
// MI455X (gfx1250) — matrix operand order repaired and checked by reading, not compiled, not run on hardware
//
#include <hip/hip_runtime.h>
#include <hip/hip_bf16.h>
#include <math.h>

typedef __attribute__((ext_vector_type(16))) _Float16 v16h;
typedef __attribute__((ext_vector_type(8)))  _Float16 v8h;
typedef __attribute__((ext_vector_type(4)))  _Float16 v4h;
typedef __attribute__((ext_vector_type(8)))  float    v8f;

#define CDIM 64
#define GDIM 256
#define RDIM 8
#define EPSV 1e-5f

union U16h { v16h v; _Float16 e[16]; };
union U8f  { v8f  v; float    e[8];  };

// ---------------------------------------------------------------- zero scratch
__global__ void k_zero(float* __restrict__ p, long long n) {
  long long i = (long long)blockIdx.x * blockDim.x + threadIdx.x;
  long long stride = (long long)gridDim.x * blockDim.x;
  for (; i < n; i += stride) p[i] = 0.f;
}

// ------------------------------------------------- edge scatter: sum + degree
// 16 threads per edge, float4 per thread; x rows + agg atomics stay in L2.
__global__ void k_scatter(const float* __restrict__ x, const int* __restrict__ ei,
                          float* __restrict__ agg, float* __restrict__ deg, int E) {
  long long gt = (long long)blockIdx.x * blockDim.x + threadIdx.x;
  int e    = (int)(gt >> 4);
  int part = (int)(gt & 15);
  if (e >= E) return;
  // prefetch a future chunk of the edge list (speculative, faults dropped)
  __builtin_prefetch(ei + e + 4096, 0, 0);
  int src = ei[e];
  int dst = ei[E + e];
  float4 v = ((const float4*)(x + (long long)src * CDIM))[part];
  float* o = agg + (long long)dst * CDIM + part * 4;
  atomicAdd(o + 0, v.x);
  atomicAdd(o + 1, v.y);
  atomicAdd(o + 2, v.z);
  atomicAdd(o + 3, v.w);
  if (part == 0) atomicAdd(deg + dst, 1.0f);
}

// ------------------------------------------------------- per-graph node count
__global__ void k_cnt(const int* __restrict__ batch, float* __restrict__ cnt, int n) {
  int i = blockIdx.x * blockDim.x + threadIdx.x;
  if (i < n) atomicAdd(&cnt[batch[i]], 1.0f);
}

// --------------------------------------------- fused WMMA node transform
// h = (agg/deg) @ Wl^T + bl + x @ Wr^T ; also accumulate per-graph sum of h.
// Block: 256 threads = 8 waves; each wave owns a 16-node row tile (128/block).
// All LDS buffers are stored in WMMA *fragment order*: each fragment is a
// contiguous 32B chunk per lane -> loads are 2x ds_load_b128, no repacking.
__global__ void __launch_bounds__(256)
k_transform(const float* __restrict__ x, const float* __restrict__ agg,
            const float* __restrict__ deg, const float* __restrict__ Wl,
            const float* __restrict__ bl, const float* __restrict__ Wr,
            const int* __restrict__ batch, float* __restrict__ h,
            float* __restrict__ sum_h, int n) {
  // activation fragments: 8 waves x 2 kc x 32 lanes x 16 halves
  __shared__ __align__(32) _Float16 sXf[8 * 2 * 32 * 16];
  __shared__ __align__(32) _Float16 sAf[8 * 2 * 32 * 16];
  // weight fragments: 4 tiles x 2 kc x 32 lanes x 16 halves
  __shared__ __align__(32) _Float16 sWlf[4 * 2 * 32 * 16];
  __shared__ __align__(32) _Float16 sWrf[4 * 2 * 32 * 16];

  const int t = threadIdx.x;
  const int base = blockIdx.x * 128;

  // ---- stage weights: one fragment per thread (256 fragments total).
  // Fragment slot (tile,kc,lane,h) needs W[o=tile*16+mn][kc+grp*16+h]:
  // h spans 16 consecutive k -> 16 contiguous floats of row o.
  {
    int tile = t >> 6;
    int kcid = (t >> 5) & 1;
    int lane = t & 31;
    int grp = lane >> 4, mn = lane & 15;
    int o = tile * 16 + mn;
    int kb = kcid * 32 + grp * 16;
    const float4* pl = (const float4*)(Wl + o * 64 + kb);
    const float4* pr = (const float4*)(Wr + o * 64 + kb);
    _Float16* dl = sWlf + t * 16;   // ((tile*2+kcid)*32+lane)*16 == t*16
    _Float16* dr = sWrf + t * 16;
#pragma unroll
    for (int q = 0; q < 4; ++q) {
      float4 a = pl[q];
      float4 b = pr[q];
      v4h ha, hb;
      ha[0] = (_Float16)a.x; ha[1] = (_Float16)a.y;
      ha[2] = (_Float16)a.z; ha[3] = (_Float16)a.w;
      hb[0] = (_Float16)b.x; hb[1] = (_Float16)b.y;
      hb[2] = (_Float16)b.z; hb[3] = (_Float16)b.w;
      *(v4h*)(dl + q * 4) = ha;
      *(v4h*)(dr + q * 4) = hb;
    }
  }

  // ---- stage 128 activation rows directly in A-fragment order.
  // For channel run c=4q..4q+3 (within 32-chunk cc): grp=cc>>4,
  // h = cc&15: 4 consecutive halves -> one 8B LDS store.
  for (int i = t; i < 128 * 16; i += 256) {
    int r = i >> 4;          // local row
    int q = i & 15;          // float4 index within row
    int node = base + r;
    float4 xv = make_float4(0.f, 0.f, 0.f, 0.f);
    float4 av = xv;
    if (node < n) {
      float d = deg[node];
      d = d < 1.f ? 1.f : d;
      float inv = 1.0f / d;
      xv = ((const float4*)(x + (long long)node * CDIM))[q];
      float4 s = ((const float4*)(agg + (long long)node * CDIM))[q];
      av = make_float4(s.x * inv, s.y * inv, s.z * inv, s.w * inv);
    }
    int w = r >> 4, m = r & 15;
    int kcid = q >> 3;
    int cc = (q & 7) * 4;                  // channel offset within 32-chunk
    int grp = cc >> 4;
    int hh = cc & 15;                      // starting half within fragment
    int off = (((w * 2 + kcid) * 32) + grp * 16 + m) * 16 + hh;
    v4h hx, ha;
    hx[0] = (_Float16)xv.x; hx[1] = (_Float16)xv.y;
    hx[2] = (_Float16)xv.z; hx[3] = (_Float16)xv.w;
    ha[0] = (_Float16)av.x; ha[1] = (_Float16)av.y;
    ha[2] = (_Float16)av.z; ha[3] = (_Float16)av.w;
    *(v4h*)(sXf + off) = hx;
    *(v4h*)(sAf + off) = ha;
  }
  __syncthreads();

  const int wave = t >> 5, lane = t & 31;
  const int grp = lane >> 4, mn = lane & 15;
  const int rowbase = wave * 16;

  auto loadFrag = [&](const _Float16* fr, int blk) {
    U16h u;
    u.v = *(const v16h*)(fr + (blk * 32 + lane) * 16);
    return u.v;
  };

  v16h aA0 = loadFrag(sAf, wave * 2 + 0);
  v16h aA1 = loadFrag(sAf, wave * 2 + 1);
  v16h aX0 = loadFrag(sXf, wave * 2 + 0);
  v16h aX1 = loadFrag(sXf, wave * 2 + 1);

  v8f acc[4] = {};
#pragma unroll
  for (int tile = 0; tile < 4; ++tile) {
    v16h bl0 = loadFrag(sWlf, tile * 2 + 0);
    v16h bl1 = loadFrag(sWlf, tile * 2 + 1);
    acc[tile] = __builtin_amdgcn_wmma_f32_16x16x32_f16(false, aA0, false, bl0,
                                                       (short)0, acc[tile], false, false);
    acc[tile] = __builtin_amdgcn_wmma_f32_16x16x32_f16(false, aA1, false, bl1,
                                                       (short)0, acc[tile], false, false);
    v16h br0 = loadFrag(sWrf, tile * 2 + 0);
    v16h br1 = loadFrag(sWrf, tile * 2 + 1);
    acc[tile] = __builtin_amdgcn_wmma_f32_16x16x32_f16(false, aX0, false, br0,
                                                       (short)0, acc[tile], false, false);
    acc[tile] = __builtin_amdgcn_wmma_f32_16x16x32_f16(false, aX1, false, br1,
                                                       (short)0, acc[tile], false, false);
  }

  // C/D layout: lane L, VGPR r -> M = r + 8*(L/16), N = L%16
#pragma unroll
  for (int tile = 0; tile < 4; ++tile) {
    U8f u; u.v = acc[tile];
    int ch = tile * 16 + mn;
    float bias = bl[ch];
#pragma unroll
    for (int r = 0; r < 8; ++r) {
      int m = r + 8 * grp;
      int node = base + rowbase + m;
      if (node < n) {
        float val = u.e[r] + bias;
        h[(long long)node * 64 + ch] = val;
        atomicAdd(&sum_h[batch[node] * 64 + ch], val);
      }
    }
  }
}

// ----------------------------------------- channel attention gate: gx = gate*x
__global__ void __launch_bounds__(256)
k_gate(const float* __restrict__ x, const float* __restrict__ a1w,
       const float* __restrict__ a1b, const float* __restrict__ a2w,
       const float* __restrict__ a2b, float* __restrict__ gx, int n) {
  __shared__ float s1[RDIM * CDIM];
  __shared__ float s2[CDIM * RDIM];
  __shared__ float sb1[RDIM];
  __shared__ float sb2[CDIM];
  int t = threadIdx.x;
  for (int i = t; i < RDIM * CDIM; i += 256) { s1[i] = a1w[i]; s2[i] = a2w[i]; }
  if (t < RDIM) sb1[t] = a1b[t];
  if (t < CDIM) sb2[t] = a2b[t];
  __syncthreads();
  int node = blockIdx.x * 256 + t;
  if (node >= n) return;
  float xr[CDIM];
  const float4* xp = (const float4*)(x + (long long)node * CDIM);
#pragma unroll
  for (int q = 0; q < 16; ++q) {
    float4 v = xp[q];
    xr[q * 4 + 0] = v.x; xr[q * 4 + 1] = v.y;
    xr[q * 4 + 2] = v.z; xr[q * 4 + 3] = v.w;
  }
  float hid[RDIM];
#pragma unroll
  for (int r = 0; r < RDIM; ++r) {
    float a = sb1[r];
#pragma unroll
    for (int k = 0; k < CDIM; ++k) a += s1[r * CDIM + k] * xr[k];
    hid[r] = a > 0.f ? a : 0.f;
  }
  float4* op = (float4*)(gx + (long long)node * CDIM);
#pragma unroll
  for (int q = 0; q < 16; ++q) {
    float4 o;
    float* oo = (float*)&o;
#pragma unroll
    for (int j = 0; j < 4; ++j) {
      int c = q * 4 + j;
      float a = sb2[c];
#pragma unroll
      for (int r = 0; r < RDIM; ++r) a += s2[c * RDIM + r] * hid[r];
      float gate = 1.0f / (1.0f + __expf(-a));
      oo[j] = gate * xr[c];
    }
    op[q] = o;
  }
}

// -------------------------------------------------------------- mean finalize
__global__ void k_mean(const float* __restrict__ sum_h, const float* __restrict__ cnt,
                       float* __restrict__ mean) {
  int i = blockIdx.x * blockDim.x + threadIdx.x;
  if (i < GDIM * CDIM) {
    float c = cnt[i >> 6];
    mean[i] = sum_h[i] / (c < 1.f ? 1.f : c);
  }
}

// ----------------------------- center (in place) + per-graph variance accum
__global__ void k_center(float* __restrict__ h, const int* __restrict__ batch,
                         const float* __restrict__ mean, const float* __restrict__ alpha,
                         float* __restrict__ sumsq, int n) {
  long long i = (long long)blockIdx.x * blockDim.x + threadIdx.x;
  if (i >= (long long)n * 16) return;
  int node = (int)(i >> 4);
  int q = (int)(i & 15);
  int g = batch[node];
  int cb = q * 4;
  float4 hv = ((float4*)h)[i];
  float4 mv = *(const float4*)(mean + g * CDIM + cb);
  float4 av = *(const float4*)(alpha + cb);
  hv.x -= mv.x * av.x; hv.y -= mv.y * av.y;
  hv.z -= mv.z * av.z; hv.w -= mv.w * av.w;
  ((float4*)h)[i] = hv;
  float* ss = sumsq + g * CDIM + cb;
  atomicAdd(ss + 0, hv.x * hv.x);
  atomicAdd(ss + 1, hv.y * hv.y);
  atomicAdd(ss + 2, hv.z * hv.z);
  atomicAdd(ss + 3, hv.w * hv.w);
}

// ------------------------------------------------------------- rstd finalize
__global__ void k_rstd(const float* __restrict__ sumsq, const float* __restrict__ cnt,
                       float* __restrict__ rstd) {
  int i = blockIdx.x * blockDim.x + threadIdx.x;
  if (i < GDIM * CDIM) {
    float c = cnt[i >> 6];
    c = c < 1.f ? 1.f : c;
    rstd[i] = rsqrtf(sumsq[i] / c + EPSV);
  }
}

// ------------------------------------ final: relu(w*ĥ*rstd + b + gate*x)
__global__ void k_final(const float* __restrict__ h, const float* __restrict__ gx,
                        const int* __restrict__ batch, const float* __restrict__ rstd,
                        const float* __restrict__ gnw, const float* __restrict__ gnb,
                        float* __restrict__ out, int n) {
  long long i = (long long)blockIdx.x * blockDim.x + threadIdx.x;
  if (i >= (long long)n * 16) return;
  int node = (int)(i >> 4);
  int q = (int)(i & 15);
  int g = batch[node];
  int cb = q * 4;
  float4 hv = ((const float4*)h)[i];
  float4 gv = ((const float4*)gx)[i];
  float4 rv = *(const float4*)(rstd + g * CDIM + cb);
  float4 wv = *(const float4*)(gnw + cb);
  float4 bv = *(const float4*)(gnb + cb);
  float4 o;
  o.x = wv.x * hv.x * rv.x + bv.x + gv.x;
  o.y = wv.y * hv.y * rv.y + bv.y + gv.y;
  o.z = wv.z * hv.z * rv.z + bv.z + gv.z;
  o.w = wv.w * hv.w * rv.w + bv.w + gv.w;
  o.x = o.x > 0.f ? o.x : 0.f;
  o.y = o.y > 0.f ? o.y : 0.f;
  o.z = o.z > 0.f ? o.z : 0.f;
  o.w = o.w > 0.f ? o.w : 0.f;
  ((float4*)out)[i] = o;
}

extern "C" void kernel_launch(void* const* d_in, const int* in_sizes, int n_in,
                              void* d_out, int out_size, void* d_ws, size_t ws_size,
                              hipStream_t stream) {
  const float* x   = (const float*)d_in[0];
  const int*  ei   = (const int*)d_in[1];
  const int*  batch= (const int*)d_in[2];
  const float* Wl  = (const float*)d_in[3];
  const float* bl  = (const float*)d_in[4];
  const float* Wr  = (const float*)d_in[5];
  const float* gnw = (const float*)d_in[6];
  const float* gnb = (const float*)d_in[7];
  const float* gna = (const float*)d_in[8];
  const float* a1w = (const float*)d_in[9];
  const float* a1b = (const float*)d_in[10];
  const float* a2w = (const float*)d_in[11];
  const float* a2b = (const float*)d_in[12];
  float* out = (float*)d_out;

  const int n = in_sizes[0] / CDIM;
  const int E = in_sizes[1] / 2;

  float* ws = (float*)d_ws;
  const long long NC = (long long)n * CDIM;
  const long long GC = (long long)GDIM * CDIM;
  float* h     = ws;                 // [N,C] -> centered in place
  float* agg   = ws + NC;            // [N,C] neighbor sums; reused as gx later
  float* deg   = ws + 2 * NC;        // [N]
  float* sum_h = deg + n;            // [G,C]
  float* sumsq = sum_h + GC;         // [G,C]
  float* cnt   = sumsq + GC;         // [G]
  float* mean  = cnt + GDIM;         // [G,C]
  float* rstd  = mean + GC;          // [G,C]

  // zero the accumulator span: agg, deg, sum_h, sumsq, cnt (contiguous)
  long long zn = NC + n + 2 * GC + GDIM;
  k_zero<<<1024, 256, 0, stream>>>(agg, zn);

  long long sth = (long long)E * 16;
  k_scatter<<<(int)((sth + 255) / 256), 256, 0, stream>>>(x, ei, agg, deg, E);
  k_cnt<<<(n + 255) / 256, 256, 0, stream>>>(batch, cnt, n);

  k_transform<<<(n + 127) / 128, 256, 0, stream>>>(x, agg, deg, Wl, bl, Wr,
                                                   batch, h, sum_h, n);

  float* gx = agg;  // agg fully consumed by k_transform
  k_gate<<<(n + 255) / 256, 256, 0, stream>>>(x, a1w, a1b, a2w, a2b, gx, n);

  k_mean<<<(GDIM * CDIM + 255) / 256, 256, 0, stream>>>(sum_h, cnt, mean);

  long long ncv = (long long)n * 16;
  k_center<<<(int)((ncv + 255) / 256), 256, 0, stream>>>(h, batch, mean, gna, sumsq, n);
  k_rstd<<<(GDIM * CDIM + 255) / 256, 256, 0, stream>>>(sumsq, cnt, rstd);
  k_final<<<(int)((ncv + 255) / 256), 256, 0, stream>>>(h, gx, batch, rstd,
                                                        gnw, gnb, out, n);
}
